// TrainableVarRegBlock_38895223833086
// MI455X (gfx1250) — matrix operand order repaired and checked by reading, not compiled, not run on hardware
//
#include <hip/hip_runtime.h>
#include <hip/hip_bf16.h>
#include <math.h>

typedef __attribute__((ext_vector_type(16))) _Float16 v16h;
typedef __attribute__((ext_vector_type(8)))  float    v8f;

#define TAU_C   1.0f
#define EPS_C   1e-6f

// gaussian sigma=1, radius=2, normalized
#define G0 0.0544886845f
#define G1 0.2442013420f
#define G2 0.4026199469f

// ---------------------------------------------------------------------------
// elementwise helpers
// ---------------------------------------------------------------------------
__global__ void zero_kernel(float* p, int n) {
    int i = blockIdx.x * blockDim.x + threadIdx.x;
    if (i < n) p[i] = 0.0f;
}

__global__ void zero_h_kernel(_Float16* p, int n) {
    int i = blockIdx.x * blockDim.x + threadIdx.x;
    if (i < n) p[i] = (_Float16)0.0f;
}

__global__ void add_kernel(const float* __restrict__ a, const float* __restrict__ b,
                           float* __restrict__ o, int n) {
    int i = blockIdx.x * blockDim.x + threadIdx.x;
    if (i < n) o[i] = a[i] + b[i];
}

// ---------------------------------------------------------------------------
// trilinear resize (half-pixel centers, edge clamp), optional output scale
// ---------------------------------------------------------------------------
__global__ void resize_tri_kernel(const float* __restrict__ src, float* __restrict__ dst,
                                  int C, int iD, int iH, int iW,
                                  int oD, int oH, int oW, float mul) {
    int oV = oD * oH * oW;
    int idx = blockIdx.x * blockDim.x + threadIdx.x;
    if (idx >= C * oV) return;
    int c = idx / oV;
    int v = idx - c * oV;
    int x = v % oW; int t = v / oW; int y = t % oH; int z = t / oH;

    float cz = (z + 0.5f) * (float)iD / oD - 0.5f;
    float cy = (y + 0.5f) * (float)iH / oH - 0.5f;
    float cx = (x + 0.5f) * (float)iW / oW - 0.5f;
    cz = fminf(fmaxf(cz, 0.0f), (float)(iD - 1));
    cy = fminf(fmaxf(cy, 0.0f), (float)(iH - 1));
    cx = fminf(fmaxf(cx, 0.0f), (float)(iW - 1));
    int z0 = (int)cz, y0 = (int)cy, x0 = (int)cx;
    int z1 = min(z0 + 1, iD - 1), y1 = min(y0 + 1, iH - 1), x1 = min(x0 + 1, iW - 1);
    float fz = cz - z0, fy = cy - y0, fx = cx - x0;
    const float* s = src + (size_t)c * iD * iH * iW;
    #define AT(zz,yy,xx) s[((zz) * iH + (yy)) * iW + (xx)]
    float v00 = AT(z0,y0,x0) * (1.f-fx) + AT(z0,y0,x1) * fx;
    float v01 = AT(z0,y1,x0) * (1.f-fx) + AT(z0,y1,x1) * fx;
    float v10 = AT(z1,y0,x0) * (1.f-fx) + AT(z1,y0,x1) * fx;
    float v11 = AT(z1,y1,x0) * (1.f-fx) + AT(z1,y1,x1) * fx;
    #undef AT
    float v0 = v00 * (1.f-fy) + v01 * fy;
    float v1 = v10 * (1.f-fy) + v11 * fy;
    dst[idx] = (v0 * (1.f-fz) + v1 * fz) * mul;
}

__global__ void resize_nearest_kernel(const float* __restrict__ src, float* __restrict__ dst,
                                      int iD, int iH, int iW, int oD, int oH, int oW) {
    int oV = oD * oH * oW;
    int idx = blockIdx.x * blockDim.x + threadIdx.x;
    if (idx >= oV) return;
    int x = idx % oW; int t = idx / oW; int y = t % oH; int z = t / oH;
    int zi = min((int)((z + 0.5f) * (float)iD / oD), iD - 1);
    int yi = min((int)((y + 0.5f) * (float)iH / oH), iH - 1);
    int xi = min((int)((x + 0.5f) * (float)iW / oW), iW - 1);
    dst[idx] = src[(zi * iH + yi) * iW + xi];
}

// ---------------------------------------------------------------------------
// warp: trilinear sample (zero outside) of single-channel vol at grid+flow
// ---------------------------------------------------------------------------
__global__ void warp_kernel(const float* __restrict__ vol, const float* __restrict__ flow,
                            float* __restrict__ out, int D, int H, int W) {
    int V = D * H * W;
    int idx = blockIdx.x * blockDim.x + threadIdx.x;
    if (idx >= V) return;
    int x = idx % W; int t = idx / W; int y = t % H; int z = t / H;
    float cz = (float)z + flow[idx];
    float cy = (float)y + flow[V + idx];
    float cx = (float)x + flow[2 * V + idx];
    int z0 = (int)floorf(cz), y0 = (int)floorf(cy), x0 = (int)floorf(cx);
    float fz = cz - z0, fy = cy - y0, fx = cx - x0;
    float acc = 0.0f;
    #pragma unroll
    for (int dz = 0; dz < 2; ++dz)
    #pragma unroll
    for (int dy = 0; dy < 2; ++dy)
    #pragma unroll
    for (int dx = 0; dx < 2; ++dx) {
        int zz = z0 + dz, yy = y0 + dy, xx = x0 + dx;
        float w = (dz ? fz : 1.f - fz) * (dy ? fy : 1.f - fy) * (dx ? fx : 1.f - fx);
        if ((unsigned)zz < (unsigned)D && (unsigned)yy < (unsigned)H && (unsigned)xx < (unsigned)W)
            acc += w * vol[(zz * H + yy) * W + xx];
    }
    out[idx] = acc;
}

// ---------------------------------------------------------------------------
// demon forces + vf update (in place)
// ---------------------------------------------------------------------------
__global__ void forces_update_kernel(const float* __restrict__ m, const float* __restrict__ fixed,
                                     const float* __restrict__ mask, float* __restrict__ vf,
                                     int D, int H, int W) {
    int V = D * H * W;
    int idx = blockIdx.x * blockDim.x + threadIdx.x;
    if (idx >= V) return;
    int x = idx % W; int t = idx / W; int y = t % H; int z = t / H;
    #define M(zz,yy,xx) m[((zz) * H + (yy)) * W + (xx)]
    float gz = (z == 0)     ? M(1,y,x) - M(0,y,x) :
               (z == D - 1) ? M(D-1,y,x) - M(D-2,y,x) : 0.5f * (M(z+1,y,x) - M(z-1,y,x));
    float gy = (y == 0)     ? M(z,1,x) - M(z,0,x) :
               (y == H - 1) ? M(z,H-1,x) - M(z,H-2,x) : 0.5f * (M(z,y+1,x) - M(z,y-1,x));
    float gx = (x == 0)     ? M(z,y,1) - M(z,y,0) :
               (x == W - 1) ? M(z,y,W-1) - M(z,y,W-2) : 0.5f * (M(z,y,x+1) - M(z,y,x-1));
    #undef M
    float diff = m[idx] - fixed[idx];
    float denom = gz * gz + gy * gy + gx * gx + diff * diff + EPS_C;
    float s = TAU_C * mask[idx] * diff / denom;
    vf[idx]         -= gz * s;
    vf[V + idx]     -= gy * s;
    vf[2 * V + idx] -= gx * s;
}

// ---------------------------------------------------------------------------
// separable gaussian, one axis, zero ('SAME') padding
// ---------------------------------------------------------------------------
__global__ void gauss_kernel(const float* __restrict__ src, float* __restrict__ dst,
                             int C, int D, int H, int W, int axis) {
    int V = D * H * W;
    int idx = blockIdx.x * blockDim.x + threadIdx.x;
    if (idx >= C * V) return;
    int c = idx / V;
    int v = idx - c * V;
    int x = v % W; int t = v / W; int y = t % H; int z = t / H;
    const float g[5] = {G0, G1, G2, G1, G0};
    float acc = 0.0f;
    #pragma unroll
    for (int k = -2; k <= 2; ++k) {
        int zz = z + (axis == 0 ? k : 0);
        int yy = y + (axis == 1 ? k : 0);
        int xx = x + (axis == 2 ? k : 0);
        if ((unsigned)zz < (unsigned)D && (unsigned)yy < (unsigned)H && (unsigned)xx < (unsigned)W)
            acc += g[k + 2] * src[c * V + (zz * H + yy) * W + xx];
    }
    dst[idx] = acc;
}

// ---------------------------------------------------------------------------
// build 5-channel f16 CNN input INTO PADDED (D+2)^3 volume: vf(3), img*mask, warped
// ---------------------------------------------------------------------------
__global__ void stack_pad_kernel(const float* __restrict__ vf, const float* __restrict__ img,
                                 const float* __restrict__ mask, const float* __restrict__ warped,
                                 _Float16* __restrict__ outp, int D, int H, int W) {
    int V = D * H * W;
    int i = blockIdx.x * blockDim.x + threadIdx.x;
    if (i >= V) return;
    int Wp = W + 2, Hp = H + 2;
    int Vp = (D + 2) * Hp * Wp;
    int x = i % W; int t = i / W; int y = t % H; int z = t / H;
    int pi = ((z + 1) * Hp + (y + 1)) * Wp + (x + 1);
    outp[pi]          = (_Float16)vf[i];
    outp[Vp + pi]     = (_Float16)vf[V + i];
    outp[2 * Vp + pi] = (_Float16)vf[2 * V + i];
    outp[3 * Vp + pi] = (_Float16)(img[i] * mask[i]);
    outp[4 * Vp + pi] = (_Float16)warped[i];
}

// ---------------------------------------------------------------------------
// weight prep: float (Cout,Cin,3,3,3) -> f16 [Npad][Cin*32], taps 27..31 and
// n >= Cout zero-padded so the conv hot loop needs no guards.
// ---------------------------------------------------------------------------
__global__ void prep_weights_kernel(const float* __restrict__ w, _Float16* __restrict__ wtp,
                                    int Cout, int Cin, int Npad) {
    int Kp = Cin * 32;
    int idx = blockIdx.x * blockDim.x + threadIdx.x;
    if (idx >= Npad * Kp) return;
    int n = idx / Kp;
    int r = idx - n * Kp;
    int cin = r >> 5;
    int tap = r & 31;
    float v = 0.0f;
    if (n < Cout && tap < 27) v = w[(n * Cin + cin) * 27 + tap];
    wtp[idx] = (_Float16)v;
}

// ---------------------------------------------------------------------------
// implicit-GEMM 3x3x3 conv via v_wmma_f32_16x16x32_f16 on zero-padded input.
// One wave32 per (4 x 16 voxels) x 16 outchannels. K laid out as Cin chunks of
// 32 (27 taps + 5 zero slots). All tap offsets are compile-time constants.
// Gather tables hold BYTE offsets; each offset is laundered through an empty
// volatile asm so the i64 zext stays adjacent to its load and ISel folds it
// into saddr + zext(voffset32) addressing (SGPR base bumped per chunk, VGPR
// byte offsets reused unchanged, zero address VALU per load).
// Per chunk: 64 u16 A-loads + one 32B B-load + 4 WMMAs.
// Pad slots (tap>=27) read channel byte 0 (zero halo).
// A frag: lane row = lane&15; slot(i,s): k = half*16 + i*2 + s.
// B frag: lane col = lane&15, element j: k = half*16 + j
// (contiguous -> single 32B load). D: row r + 8*half, col lane&15.
// ---------------------------------------------------------------------------
#define MT 4

template <int D, int H, int W>
__global__ __launch_bounds__(32)
void conv3d_wmma_kernel(const _Float16* __restrict__ in16,  // [Cin][Vp] padded
                        const _Float16* __restrict__ wtp,   // [Npad][Cin*32]
                        float* __restrict__ out,            // [Cout][V]
                        const float* __restrict__ bias,     // optional
                        int Cin, int Cout) {
    constexpr int Wp = W + 2, Hp = H + 2, Dp = D + 2;
    constexpr int Vp = Dp * Hp * Wp;
    constexpr int V  = D * H * W;
    const int Kp  = Cin * 32;
    const int lane = threadIdx.x;
    const int row  = lane & 15;
    const int half = lane >> 4;
    const int nCol = blockIdx.y * 16 + row;
    const int tile0 = blockIdx.x * (16 * MT);

    // per-lane BYTE-offset gather tables (channel-relative)
    unsigned aoff[MT][16];
    #pragma unroll
    for (int mt = 0; mt < MT; ++mt) {
        int vox = tile0 + mt * 16 + row;
        int x = vox % W; int t = vox / W; int y = t % H; int z = t / H;
        int base = ((z + 1) * Hp + (y + 1)) * Wp + (x + 1);
        #pragma unroll
        for (int i = 0; i < 8; ++i) {
            #pragma unroll
            for (int s = 0; s < 2; ++s) {
                const int kw = i * 2 + s;  // compile-time
                int tap = kw + half * 16;                            // half: 0/1 runtime
                int dz = tap / 9;
                int r9 = tap - dz * 9;
                int dy = r9 / 3;
                int dx = r9 - dy * 3;
                int off = (dz - 1) * (Hp * Wp) + (dy - 1) * Wp + (dx - 1);
                aoff[mt][i * 2 + s] = (tap < 27) ? (unsigned)((base + off) * 2) : 0u;
            }
        }
    }

    v8f acc[MT];
    #pragma unroll
    for (int mt = 0; mt < MT; ++mt)
        acc[mt] = (v8f){0.f, 0.f, 0.f, 0.f, 0.f, 0.f, 0.f, 0.f};

    const unsigned boff = (unsigned)((nCol * Kp + half * 16) * 2);  // bytes, per-lane
    const char* aBase = (const char*)in16;  // uniform, advances by Vp*2 per chunk
    const char* bBase = (const char*)wtp;   // uniform, advances by 64 per chunk

    for (int cin = 0; cin < Cin; ++cin) {
        unsigned bo = boff;
        asm volatile("" : "+v"(bo));        // keep zext local to the load
        v16h b = *(const v16h*)(bBase + bo);
        #pragma unroll
        for (int mt = 0; mt < MT; ++mt) {
            v16h a;
            #pragma unroll
            for (int sl = 0; sl < 16; ++sl) {
                unsigned o = aoff[mt][sl];
                asm volatile("" : "+v"(o)); // keep zext local to the load
                a[sl] = *(const _Float16*)(aBase + o);
            }
            acc[mt] = __builtin_amdgcn_wmma_f32_16x16x32_f16(
                false, a, false, b, (short)0, acc[mt], false, false);
        }
        aBase += (size_t)(Vp * 2);
        bBase += 64;
    }

    if (nCol < Cout) {
        float bi = bias ? bias[nCol] : 0.0f;
        #pragma unroll
        for (int mt = 0; mt < MT; ++mt) {
            #pragma unroll
            for (int r = 0; r < 8; ++r) {
                int vox = tile0 + mt * 16 + r + half * 8;
                out[nCol * V + vox] = acc[mt][r] + bi;
            }
        }
    }
}

// ---------------------------------------------------------------------------
// per-channel mean/var (one workgroup per channel)
// ---------------------------------------------------------------------------
__global__ void stats_kernel(const float* __restrict__ x, float* __restrict__ stats, int V) {
    __shared__ float ss[256], s2[256];
    int c = blockIdx.x;
    float a = 0.f, b = 0.f;
    for (int i = threadIdx.x; i < V; i += blockDim.x) {
        float v = x[(size_t)c * V + i];
        a += v; b += v * v;
    }
    ss[threadIdx.x] = a; s2[threadIdx.x] = b;
    __syncthreads();
    for (int o = 128; o > 0; o >>= 1) {
        if (threadIdx.x < o) { ss[threadIdx.x] += ss[threadIdx.x + o]; s2[threadIdx.x] += s2[threadIdx.x + o]; }
        __syncthreads();
    }
    if (threadIdx.x == 0) {
        float mean = ss[0] / V;
        stats[2 * c] = mean;
        stats[2 * c + 1] = s2[0] / V - mean * mean;
    }
}

// instance-norm + mish, write f16 into PADDED volume (feeds next WMMA layer)
__global__ void norm_mish_pad_kernel(const float* __restrict__ x, const float* __restrict__ stats,
                                     _Float16* __restrict__ outp, int C, int D, int H, int W) {
    int V = D * H * W;
    int idx = blockIdx.x * blockDim.x + threadIdx.x;
    if (idx >= C * V) return;
    int c = idx / V;
    int v = idx - c * V;
    int Wp = W + 2, Hp = H + 2;
    int Vp = (D + 2) * Hp * Wp;
    int xx = v % W; int t = v / W; int yy = t % H; int zz = t / H;
    float mean = stats[2 * c], var = stats[2 * c + 1];
    float y = (x[idx] - mean) * rsqrtf(var + 1e-5f);
    float sp = (y > 20.0f) ? y : log1pf(expf(y));
    outp[c * Vp + ((zz + 1) * Hp + (yy + 1)) * Wp + (xx + 1)] = (_Float16)(y * tanhf(sp));
}

// ---------------------------------------------------------------------------
// host orchestration
// ---------------------------------------------------------------------------
extern "C" void kernel_launch(void* const* d_in, const int* in_sizes, int n_in,
                              void* d_out, int out_size, void* d_ws, size_t ws_size,
                              hipStream_t stream) {
    (void)in_sizes; (void)n_in; (void)out_size; (void)ws_size;
    const float* img  = (const float*)d_in[0];
    const float* mask = (const float*)d_in[1];
    const float* cond = (const float*)d_in[2];
    const float* w1   = (const float*)d_in[3];
    const float* w2   = (const float*)d_in[4];
    const float* w3   = (const float*)d_in[5];
    const float* w4   = (const float*)d_in[6];
    const float* b4   = (const float*)d_in[7];
    float* out = (float*)d_out;

    const int D = 96, H = 96, W = 96;    const int V  = D * H * W;        // 884736
    const int D1 = 48, H1 = 48, W1 = 48; const int V1 = D1 * H1 * W1;     // 110592
    const int Vp = (D + 2) * (H + 2) * (W + 2);                           // 941192
    const int TPB = 256;
    auto blk = [](int n, int t) { return (n + t - 1) / t; };

    // workspace layout
    size_t off = 0;
    char* wsb = (char*)d_ws;
    auto Af = [&](size_t n) -> float* {
        float* p = (float*)(wsb + off); off += ((n * 4 + 255) & ~(size_t)255); return p; };
    auto Ah = [&](size_t n) -> _Float16* {
        _Float16* p = (_Float16*)(wsb + off); off += ((n * 2 + 255) & ~(size_t)255); return p; };

    float* img48   = Af(V1);
    float* cond48  = Af(V1);
    float* mask48  = Af(V1);
    float* vf48    = Af(3 * (size_t)V1);
    float* t48a    = Af(3 * (size_t)V1);
    float* t48b    = Af(3 * (size_t)V1);
    float* warp48  = Af(V1);
    float* vf96    = Af(3 * (size_t)V);
    float* t96a    = Af(3 * (size_t)V);
    float* t96b    = Af(3 * (size_t)V);
    float* warp96  = Af(V);
    float* corr    = Af(3 * (size_t)V);    // also reused as final cvf
    float* stats   = Af(128);
    _Float16* stk16p = Ah(5 * (size_t)Vp);          // padded 5-ch CNN input
    _Float16* w1p  = Ah(32 * (5  * 32));
    _Float16* w2p  = Ah(64 * (32 * 32));
    _Float16* w3p  = Ah(32 * (64 * 32));
    _Float16* w4p  = Ah(16 * (32 * 32));
    float*    convf = Af(64 * (size_t)V);           // raw conv output, reused
    _Float16* a16p  = Ah(64 * (size_t)Vp);          // padded normalized f16 acts

    // ---- level 0: 48^3, 4 iterations ----
    resize_tri_kernel<<<blk(V1, TPB), TPB, 0, stream>>>(img,  img48,  1, D, H, W, D1, H1, W1, 1.0f);
    resize_tri_kernel<<<blk(V1, TPB), TPB, 0, stream>>>(cond, cond48, 1, D, H, W, D1, H1, W1, 1.0f);
    resize_nearest_kernel<<<blk(V1, TPB), TPB, 0, stream>>>(mask, mask48, D, H, W, D1, H1, W1);
    zero_kernel<<<blk(3 * V1, TPB), TPB, 0, stream>>>(vf48, 3 * V1);
    for (int it = 0; it < 4; ++it) {
        warp_kernel<<<blk(V1, TPB), TPB, 0, stream>>>(cond48, vf48, warp48, D1, H1, W1);
        forces_update_kernel<<<blk(V1, TPB), TPB, 0, stream>>>(warp48, img48, mask48, vf48, D1, H1, W1);
        gauss_kernel<<<blk(3 * V1, TPB), TPB, 0, stream>>>(vf48, t48a, 3, D1, H1, W1, 0);
        gauss_kernel<<<blk(3 * V1, TPB), TPB, 0, stream>>>(t48a, t48b, 3, D1, H1, W1, 1);
        gauss_kernel<<<blk(3 * V1, TPB), TPB, 0, stream>>>(t48b, vf48, 3, D1, H1, W1, 2);
    }

    // ---- level 1: 96^3, 4 iterations ----
    resize_tri_kernel<<<blk(3 * V, TPB), TPB, 0, stream>>>(vf48, vf96, 3, D1, H1, W1, D, H, W, 2.0f);
    for (int it = 0; it < 4; ++it) {
        warp_kernel<<<blk(V, TPB), TPB, 0, stream>>>(cond, vf96, warp96, D, H, W);
        forces_update_kernel<<<blk(V, TPB), TPB, 0, stream>>>(warp96, img, mask, vf96, D, H, W);
        gauss_kernel<<<blk(3 * V, TPB), TPB, 0, stream>>>(vf96, t96a, 3, D, H, W, 0);
        gauss_kernel<<<blk(3 * V, TPB), TPB, 0, stream>>>(t96a, t96b, 3, D, H, W, 1);
        gauss_kernel<<<blk(3 * V, TPB), TPB, 0, stream>>>(t96b, vf96, 3, D, H, W, 2);
    }

    // ---- CNN ----
    // zero halos once (interiors fully rewritten each layer)
    zero_h_kernel<<<blk(5 * Vp, TPB), TPB, 0, stream>>>(stk16p, 5 * Vp);
    zero_h_kernel<<<blk(64 * Vp, TPB), TPB, 0, stream>>>(a16p, 64 * Vp);
    stack_pad_kernel<<<blk(V, TPB), TPB, 0, stream>>>(vf96, img, mask, warp96, stk16p, D, H, W);
    prep_weights_kernel<<<blk(32 * 5 * 32, TPB), TPB, 0, stream>>>(w1, w1p, 32, 5, 32);
    prep_weights_kernel<<<blk(64 * 32 * 32, TPB), TPB, 0, stream>>>(w2, w2p, 64, 32, 64);
    prep_weights_kernel<<<blk(32 * 64 * 32, TPB), TPB, 0, stream>>>(w3, w3p, 32, 64, 32);
    prep_weights_kernel<<<blk(16 * 32 * 32, TPB), TPB, 0, stream>>>(w4, w4p, 3, 32, 16);

    dim3 wave(32, 1, 1);
    const int gx = V / (16 * MT);   // 13824
    // conv1: 5 -> 32
    conv3d_wmma_kernel<96, 96, 96><<<dim3(gx, 2), wave, 0, stream>>>(stk16p, w1p, convf, nullptr, 5, 32);
    stats_kernel<<<32, 256, 0, stream>>>(convf, stats, V);
    norm_mish_pad_kernel<<<blk(32 * V, TPB), TPB, 0, stream>>>(convf, stats, a16p, 32, D, H, W);
    // conv2: 32 -> 64
    conv3d_wmma_kernel<96, 96, 96><<<dim3(gx, 4), wave, 0, stream>>>(a16p, w2p, convf, nullptr, 32, 64);
    stats_kernel<<<64, 256, 0, stream>>>(convf, stats, V);
    norm_mish_pad_kernel<<<blk(64 * V, TPB), TPB, 0, stream>>>(convf, stats, a16p, 64, D, H, W);
    // conv3: 64 -> 32
    conv3d_wmma_kernel<96, 96, 96><<<dim3(gx, 2), wave, 0, stream>>>(a16p, w3p, convf, nullptr, 64, 32);
    stats_kernel<<<32, 256, 0, stream>>>(convf, stats, V);
    norm_mish_pad_kernel<<<blk(32 * V, TPB), TPB, 0, stream>>>(convf, stats, a16p, 32, D, H, W);
    // conv4: 32 -> 3 (+bias)
    conv3d_wmma_kernel<96, 96, 96><<<dim3(gx, 1), wave, 0, stream>>>(a16p, w4p, corr, b4, 32, 3);

    // ---- cvf = gaussian_smooth(vf + corr) ----
    add_kernel<<<blk(3 * V, TPB), TPB, 0, stream>>>(vf96, corr, t96a, 3 * V);
    gauss_kernel<<<blk(3 * V, TPB), TPB, 0, stream>>>(t96a, t96b, 3, D, H, W, 0);
    gauss_kernel<<<blk(3 * V, TPB), TPB, 0, stream>>>(t96b, t96a, 3, D, H, W, 1);
    gauss_kernel<<<blk(3 * V, TPB), TPB, 0, stream>>>(t96a, corr, 3, D, H, W, 2);  // corr = cvf

    // ---- outputs: corrected_warped, warped_full, cvf_full, vf_full ----
    warp_kernel<<<blk(V, TPB), TPB, 0, stream>>>(cond, corr, out, D, H, W);
    warp_kernel<<<blk(V, TPB), TPB, 0, stream>>>(cond, vf96, out + V, D, H, W);
    hipMemcpyAsync(out + 2 * (size_t)V, corr, 3 * (size_t)V * sizeof(float),
                   hipMemcpyDeviceToDevice, stream);
    hipMemcpyAsync(out + 5 * (size_t)V, vf96, 3 * (size_t)V * sizeof(float),
                   hipMemcpyDeviceToDevice, stream);
}
